// VanillaMAPPOMicroCommTargetedFusionAgent_30408368455783
// MI455X (gfx1250) — matrix operand order repaired and checked by reading, not compiled, not run on hardware
//
#include <hip/hip_runtime.h>
#include <hip/hip_bf16.h>
#include <stddef.h>

typedef unsigned short u16;

// ---------------- constants (shapes fixed by the reference) ----------------
constexpr int ROWS  = 27648;        // BS * N = 1024 * 27
constexpr int NAG   = 27;
constexpr int HID   = 256;
constexpr int NA    = 33;
constexpr int OFFB  = 6;
constexpr int ATK   = 27;
constexpr int HEADS = 4;
constexpr int HD    = 16;
constexpr int VDIM  = 8;
constexpr int FLAT  = 32;
constexpr float NEGV = -1e10f;

// ---------------- workspace layout (bytes) ----------------
// R1: in_bf (27648x352 bf16) -> later reused for h_bf (27648x256 bf16)
constexpr size_t OB_INBF   = 0;
constexpr size_t OB_HBF    = 0;                                  // alias (in_bf dead after GEMM1)
constexpr size_t OB_HIDBF  = OB_INBF + (size_t)ROWS*352*2;
constexpr size_t OB_XBF    = OB_HIDBF + (size_t)ROWS*256*2;
constexpr size_t OB_W      = OB_XBF + (size_t)ROWS*256*2;        // weights bf16 (row-padded)
// weight sub-offsets (bf16 elements, all multiples of 8); rows padded to tile width
constexpr size_t WE_W1  = 0;                     // 256x352
constexpr size_t WE_WIH = WE_W1  + 256*352;      // 768x256
constexpr size_t WE_WHH = WE_WIH + 768*256;      // 768x256
constexpr size_t WE_WP1 = WE_WHH + 768*256;      // 256x256
constexpr size_t WE_WP2 = WE_WP1 + 256*256;      // 64x256  (33 rows + zero pad)
constexpr size_t WE_WV  = WE_WP2 + 64*256;       // 64x32   (32 rows + zero pad)
constexpr size_t WE_WQ  = WE_WV  + 64*32;        // 64x256
constexpr size_t WE_WKH = WE_WQ  + 64*256;       // 64x256
constexpr size_t WE_WG1 = WE_WKH + 64*256;       // 256x288
constexpr size_t WE_WD1 = WE_WG1 + 256*288;      // 256x288
constexpr size_t WE_WD2 = WE_WD1 + 256*288;      // 64x256  (27 rows + zero pad)
constexpr size_t WE_END = WE_WD2 + 64*256;
constexpr size_t OB_GI     = OB_W + WE_END*2;                    // 27648x768 f32
constexpr size_t OB_GH     = OB_GI + (size_t)ROWS*768*4;         // 27648x768 f32
// aliases inside GI region (gi dead after GRU combine)
constexpr size_t OB_LOGITS = OB_GI;                              // 27648x48 f32
constexpr size_t OB_SV     = OB_LOGITS + (size_t)ROWS*48*4;      // 27648x32 f32
constexpr size_t OB_Q      = OB_SV     + (size_t)ROWS*32*4;      // 27648x64 f32
constexpr size_t OB_KH     = OB_Q      + (size_t)ROWS*64*4;      // 27648x64 f32
constexpr size_t OB_GATE   = OB_KH     + (size_t)ROWS*64*4;      // 27648 f32
constexpr size_t OB_DELTA  = OB_GATE   + (size_t)ROWS*4;         // 27648x32 f32
constexpr size_t OB_PBF    = OB_DELTA  + (size_t)ROWS*32*4;      // 27648x256 bf16
constexpr size_t OB_VSBF   = OB_PBF    + (size_t)ROWS*256*2;     // 27648x32 bf16
constexpr size_t OB_G1BF   = OB_VSBF   + (size_t)ROWS*32*2;      // 27648x256 bf16
constexpr size_t OB_D1BF   = OB_G1BF   + (size_t)ROWS*256*2;     // 27648x256 bf16
// alias inside GH region (gh dead after GRU combine)
constexpr size_t OB_FUSBF  = OB_GH;                              // 27648x288 bf16

// ---------------- types for WMMA ----------------
typedef __attribute__((ext_vector_type(16))) __bf16        v16bf;
typedef __attribute__((ext_vector_type(8)))  float         v8f;
typedef __attribute__((ext_vector_type(4)))  unsigned int  v4u;

union Frag16 { v16bf bf; struct { v4u lo, hi; } q; };

__device__ __forceinline__ unsigned int f2bf(float x) {
    unsigned int u = __float_as_uint(x);
    u += 0x7FFFu + ((u >> 16) & 1u);          // round-to-nearest-even
    return u >> 16;
}
__device__ __forceinline__ float bf2f(u16 x) {
    return __uint_as_float((unsigned int)x << 16);
}

// ------- pad/convert f32 -> bf16 (zero pad rows srows..drows, cols scols..dld) -------
__global__ void k_cvt(const float* __restrict__ src, u16* __restrict__ dst,
                      int srows, int drows, int scols, int sld, int dld)
{
    int t = blockIdx.x * 256 + threadIdx.x;
    if (t >= drows * dld) return;
    int r = t / dld, c = t - r * dld;
    float v = (r < srows && c < scols) ? src[(size_t)r * sld + c] : 0.f;
    dst[t] = (u16)f2bf(v);
}

// ---------------- bf16 WMMA GEMM: C[M x Ncols] = act(A[M x K] @ B[Ncols x K]^T + bias) ----
// A, B bf16; lda/ldb multiples of 8; K multiple of 32; M = 27648 (exact 128-tiles);
// B row count padded to a multiple of TN (zero rows) -> no guards in the hot loop.
// tile: 128(M) x TN(N); 8 waves: 4 along M, 2 along N; each wave 32 x TN/2.
template<int ACT, bool OUTBF, int TN>   // ACT: 0 none, 1 relu; TN: 64 or 128
__global__ __launch_bounds__(256)
void gemm_bf16_kernel(const u16* __restrict__ A, const u16* __restrict__ B,
                      const float* __restrict__ bias, void* __restrict__ Cv,
                      int K, int Ncols, int lda, int ldb, int ldc)
{
    constexpr int NI    = TN / 32;      // B sub-tiles per wave (2 or 4)
    constexpr int NB_CH = TN / 64;      // B 8-elem chunks per thread (1 or 2)

    __shared__ u16 As[128][40];         // padded stride: 80B rows, frag cols 16B-aligned
    __shared__ u16 Bs[TN][40];

    const int tid   = threadIdx.x;
    const int lane  = tid & 31;
    const int wave  = tid >> 5;
    const int waveM = wave & 3;
    const int waveN = wave >> 2;
    const int l16   = lane & 15;
    const int half  = lane >> 4;
    const int mBase = blockIdx.y * 128;
    const int nBase = blockIdx.x * TN;

    v8f acc[2][NI];
#pragma unroll
    for (int mi = 0; mi < 2; ++mi)
#pragma unroll
        for (int ni = 0; ni < NI; ++ni)
            acc[mi][ni] = v8f{0.f,0.f,0.f,0.f,0.f,0.f,0.f,0.f};

    v4u arg[2], brg[NB_CH];             // prefetch regs

    auto fetch = [&](int k0) {
#pragma unroll
        for (int it = 0; it < 2; ++it) {
            int e = it * 256 + tid;     // 8-elem chunk id; 4 chunks per 32-wide row
            int r = e >> 2, c = (e & 3) * 8;
            arg[it] = *(const v4u*)(A + (size_t)(mBase + r) * lda + k0 + c);
        }
#pragma unroll
        for (int it = 0; it < NB_CH; ++it) {
            int e = it * 256 + tid;
            int r = e >> 2, c = (e & 3) * 8;
            brg[it] = *(const v4u*)(B + (size_t)(nBase + r) * ldb + k0 + c);
        }
    };
    auto stage = [&]() {
#pragma unroll
        for (int it = 0; it < 2; ++it) {
            int e = it * 256 + tid;
            int r = e >> 2, c = (e & 3) * 8;
            *(v4u*)&As[r][c] = arg[it];
        }
#pragma unroll
        for (int it = 0; it < NB_CH; ++it) {
            int e = it * 256 + tid;
            int r = e >> 2, c = (e & 3) * 8;
            *(v4u*)&Bs[r][c] = brg[it];
        }
    };

    fetch(0);
    for (int k0 = 0; k0 < K; k0 += 32) {
        __syncthreads();                       // previous compute done reading LDS
        stage();
        __syncthreads();
        if (k0 + 32 < K) fetch(k0 + 32);       // overlap global latency with WMMAs

        Frag16 af[2], bf[NI];
#pragma unroll
        for (int mi = 0; mi < 2; ++mi) {
            int row = waveM * 32 + mi * 16 + l16;
            af[mi].q.lo = *(const v4u*)&As[row][half * 8];        // K: half*8 .. +7
            af[mi].q.hi = *(const v4u*)&As[row][16 + half * 8];   // K: 16+half*8 .. +7
        }
#pragma unroll
        for (int ni = 0; ni < NI; ++ni) {
            int col = waveN * (TN / 2) + ni * 16 + l16;
            bf[ni].q.lo = *(const v4u*)&Bs[col][half * 8];        // K: half*8 .. +7
            bf[ni].q.hi = *(const v4u*)&Bs[col][16 + half * 8];   // K: 16+half*8 .. +7
        }
#pragma unroll
        for (int mi = 0; mi < 2; ++mi)
#pragma unroll
            for (int ni = 0; ni < NI; ++ni)
                acc[mi][ni] = __builtin_amdgcn_wmma_f32_16x16x32_bf16(
                    false, af[mi].bf, false, bf[ni].bf,
                    (short)0, acc[mi][ni], false, false);
    }

    // epilogue: lane holds N = lane%16, VGPR r holds M = r + (lane/16)*8
    float* Cf = (float*)Cv;
    u16*   Cb = (u16*)Cv;
#pragma unroll
    for (int ni = 0; ni < NI; ++ni) {
        int n = nBase + waveN * (TN / 2) + ni * 16 + l16;
        if (n >= Ncols) continue;
        float bv = bias ? bias[n] : 0.f;
#pragma unroll
        for (int mi = 0; mi < 2; ++mi) {
#pragma unroll
            for (int r = 0; r < 8; ++r) {
                int m = mBase + waveM * 32 + mi * 16 + r + half * 8;
                float v = acc[mi][ni][r] + bv;
                if (ACT == 1) v = v > 0.f ? v : 0.f;
                if (OUTBF) Cb[(size_t)m * ldc + n] = (u16)f2bf(v);
                else       Cf[(size_t)m * ldc + n] = v;
            }
        }
    }
}

// ---------------- GRU combine: h = (1-z)*n + z*h_in (writes f32 out + bf16 copy) ----
__global__ void k_gru(const float* __restrict__ gi, const float* __restrict__ gh,
                      const float* __restrict__ hin, float* __restrict__ h,
                      u16* __restrict__ hbf)
{
    int t = blockIdx.x * 256 + threadIdx.x;
    if (t >= ROWS * HID) return;
    int row = t >> 8, c = t & 255;
    size_t b = (size_t)row * 768;
    float ir = gi[b + c],        hr = gh[b + c];
    float iz = gi[b + 256 + c],  hz = gh[b + 256 + c];
    float in_ = gi[b + 512 + c], hn = gh[b + 512 + c];
    float r = 1.f / (1.f + __expf(-(ir + hr)));
    float z = 1.f / (1.f + __expf(-(iz + hz)));
    float n = tanhf(in_ + r * hn);
    float hv = (1.f - z) * n + z * hin[t];
    h[t] = hv;
    hbf[t] = (u16)f2bf(hv);
}

__global__ void k_fuse_h(const u16* __restrict__ hbf, u16* __restrict__ fusion)
{
    int t = blockIdx.x * 256 + threadIdx.x;
    if (t >= ROWS * HID) return;
    int row = t >> 8, c = t & 255;
    fusion[(size_t)row * 288 + c] = hbf[t];
}

// ---------------- attack softmax -> value_source (bf16, ld 32) ----------------
__global__ void k_value_source(const float* __restrict__ logits, u16* __restrict__ vs)
{
    int row = blockIdx.x * 256 + threadIdx.x;
    if (row >= ROWS) return;
    const float* al = logits + (size_t)row * 48 + OFFB;
    float m = -1e30f;
    for (int j = 0; j < ATK; ++j) m = fmaxf(m, al[j]);
    float p[ATK]; float s = 0.f;
    for (int j = 0; j < ATK; ++j) { p[j] = __expf(al[j] - m); s += p[j]; }
    float inv = 1.f / s, top = 0.f;
    u16* o = vs + (size_t)row * 32;
    for (int j = 0; j < ATK; ++j) { float q = p[j] * inv; o[j] = (u16)f2bf(q); top = fmaxf(top, q); }
    o[27] = 0x3F80;                 // 1.0f
    o[28] = (u16)f2bf(top);
    o[29] = 0; o[30] = 0; o[31] = 0;
}

// ---------------- attention core: scores, top-4 mask, softmax, messages, layernorm ----
__global__ __launch_bounds__(128)
void k_attn(const float* __restrict__ q, const float* __restrict__ kh,
            const float* __restrict__ rel, const float* __restrict__ Wk,
            const float* __restrict__ bk, const float* __restrict__ sv,
            const float* __restrict__ nullk, const float* __restrict__ nullv,
            const float* __restrict__ ln_g, const float* __restrict__ ln_b,
            u16* __restrict__ fusion)
{
    __shared__ float smsg[32][33];
    int t = blockIdx.x * 128 + threadIdx.x;       // grid exact: 864*128 == ROWS*4
    int pair = t >> 2;
    int h = t & 3;
    int b = pair / NAG, i = pair - b * NAG;
    const float scale = 0.25f;                    // 1/sqrt(16)

    float qv[HD];
#pragma unroll
    for (int d = 0; d < HD; ++d) qv[d] = q[(size_t)pair * 64 + h * 16 + d];

    float sc[28];
    {
        float s = 0.f;
#pragma unroll
        for (int d = 0; d < HD; ++d) s += qv[d] * nullk[h * 16 + d];
        sc[27] = s * scale;
    }
    const float* relbi = rel + (size_t)pair * NAG * 5;
    for (int j = 0; j < NAG; ++j) {
        const float* khr = kh + (size_t)(b * NAG + j) * 64 + h * 16;
        float s = 0.f;
#pragma unroll
        for (int d = 0; d < HD; ++d) {
            float key = khr[d] + bk[h * 16 + d];
            const float* wkr = Wk + (size_t)(h * 16 + d) * 261 + 256;
#pragma unroll
            for (int r = 0; r < 5; ++r) key += relbi[j * 5 + r] * wkr[r];
            s += qv[d] * key;
        }
        sc[j] = (j == i) ? NEGV : s * scale;
    }

    float t0 = -1e30f, t1 = -1e30f, t2 = -1e30f, t3 = -1e30f;
    for (int k2 = 0; k2 < 28; ++k2) {
        float v = sc[k2];
        if (v > t0)      { t3 = t2; t2 = t1; t1 = t0; t0 = v; }
        else if (v > t1) { t3 = t2; t2 = t1; t1 = v; }
        else if (v > t2) { t3 = t2; t2 = v; }
        else if (v > t3) { t3 = v; }
    }
    float thr = t3, m = t0, sum = 0.f;
    for (int k2 = 0; k2 < 28; ++k2) {
        float a = (sc[k2] >= thr) ? __expf(sc[k2] - m) : 0.f;
        sc[k2] = a; sum += a;
    }
    float inv = 1.f / sum;

    float msg[VDIM];
#pragma unroll
    for (int v = 0; v < VDIM; ++v) msg[v] = 0.f;
    for (int j = 0; j < NAG; ++j) {
        float a = sc[j] * inv;
        const float* svr = sv + (size_t)(b * NAG + j) * 32 + h * 8;
#pragma unroll
        for (int v = 0; v < VDIM; ++v) msg[v] += a * svr[v];
    }
    {
        float an = sc[27] * inv;
#pragma unroll
        for (int v = 0; v < VDIM; ++v) msg[v] += an * nullv[h * 8 + v];
    }

    int g = threadIdx.x >> 2;
#pragma unroll
    for (int v = 0; v < VDIM; ++v) smsg[g][h * 8 + v] = msg[v];
    __syncthreads();
    float mu = 0.f;
#pragma unroll
    for (int c = 0; c < FLAT; ++c) mu += smsg[g][c];
    mu *= (1.f / FLAT);
    float var = 0.f;
#pragma unroll
    for (int c = 0; c < FLAT; ++c) { float d = smsg[g][c] - mu; var += d * d; }
    var *= (1.f / FLAT);
    float rinv = rsqrtf(var + 1e-5f);
#pragma unroll
    for (int v = 0; v < VDIM; ++v) {
        int c = h * 8 + v;
        float o = (msg[v] - mu) * rinv * ln_g[c] + ln_b[c];
        fusion[(size_t)pair * 288 + 256 + c] = (u16)f2bf(o);
    }
}

// ---------------- gate: sigmoid(g1 @ Wg2^T + bg2) (g1 is bf16) ----------------
__global__ void k_gate(const u16* __restrict__ g1, const float* __restrict__ Wg2,
                       const float* __restrict__ bg2, float* __restrict__ gate)
{
    int row = blockIdx.x * 256 + threadIdx.x;
    if (row >= ROWS) return;
    const u16* a = g1 + (size_t)row * 256;
    float s = 0.f;
    for (int k = 0; k < 256; ++k) s += bf2f(a[k]) * Wg2[k];
    gate[row] = 1.f / (1.f + __expf(-(s + bg2[0])));
}

// ---------------- final: concat(base, attack + 0.1*gate*delta) ----------------
__global__ void k_final(const float* __restrict__ logits, const float* __restrict__ gate,
                        const float* __restrict__ delta, float* __restrict__ out)
{
    int t = blockIdx.x * 256 + threadIdx.x;
    if (t >= ROWS * NA) return;
    int row = t / NA, a = t - row * NA;
    float v = logits[(size_t)row * 48 + a];
    if (a >= OFFB) v += 0.1f * gate[row] * delta[(size_t)row * 32 + (a - OFFB)];
    out[t] = v;
}

// ---------------- host ----------------
static void launch_gemm(bool relu_bf16out, int tn, void* C, const u16* A, const u16* B,
                        const float* bias, int K, int Ncols, int lda, int ldb, int ldc,
                        hipStream_t s)
{
    dim3 grid((Ncols + tn - 1) / tn, ROWS / 128);
    if (relu_bf16out)        // all relu+bf16out GEMMs are wide (N multiple of 128)
        gemm_bf16_kernel<1, true, 128><<<grid, 256, 0, s>>>(A, B, bias, C, K, Ncols, lda, ldb, ldc);
    else if (tn == 128)
        gemm_bf16_kernel<0, false, 128><<<grid, 256, 0, s>>>(A, B, bias, C, K, Ncols, lda, ldb, ldc);
    else
        gemm_bf16_kernel<0, false, 64><<<grid, 256, 0, s>>>(A, B, bias, C, K, Ncols, lda, ldb, ldc);
}

extern "C" void kernel_launch(void* const* d_in, const int* in_sizes, int n_in,
                              void* d_out, int out_size, void* d_ws, size_t ws_size,
                              hipStream_t stream)
{
    (void)in_sizes; (void)n_in; (void)out_size; (void)ws_size;
    const float* inputs = (const float*)d_in[0];
    const float* hidden = (const float*)d_in[1];
    const float* rel    = (const float*)d_in[2];
    const float* W1  = (const float*)d_in[3];  const float* b1  = (const float*)d_in[4];
    const float* Wih = (const float*)d_in[5];  const float* bih = (const float*)d_in[6];
    const float* Whh = (const float*)d_in[7];  const float* bhh = (const float*)d_in[8];
    const float* Wp1 = (const float*)d_in[9];  const float* bp1 = (const float*)d_in[10];
    const float* Wp2 = (const float*)d_in[11]; const float* bp2 = (const float*)d_in[12];
    const float* Wq  = (const float*)d_in[13]; const float* bq  = (const float*)d_in[14];
    const float* Wk  = (const float*)d_in[15]; const float* bk  = (const float*)d_in[16];
    const float* Wv  = (const float*)d_in[17]; const float* bv  = (const float*)d_in[18];
    const float* lng = (const float*)d_in[19]; const float* lnb = (const float*)d_in[20];
    const float* Wg1 = (const float*)d_in[21]; const float* bg1 = (const float*)d_in[22];
    const float* Wg2 = (const float*)d_in[23]; const float* bg2 = (const float*)d_in[24];
    const float* Wd1 = (const float*)d_in[25]; const float* bd1 = (const float*)d_in[26];
    const float* Wd2 = (const float*)d_in[27]; const float* bd2 = (const float*)d_in[28];
    const float* nullk = (const float*)d_in[29];
    const float* nullv = (const float*)d_in[30];

    char*  wsb = (char*)d_ws;
    float* out = (float*)d_out;
    float* h   = out + (size_t)ROWS * NA;          // second tuple output (f32)

    u16* in_bf  = (u16*)(wsb + OB_INBF);
    u16* h_bf   = (u16*)(wsb + OB_HBF);
    u16* hid_bf = (u16*)(wsb + OB_HIDBF);
    u16* x_bf   = (u16*)(wsb + OB_XBF);
    u16* wbf    = (u16*)(wsb + OB_W);
    float* gi     = (float*)(wsb + OB_GI);
    float* gh     = (float*)(wsb + OB_GH);
    float* logits = (float*)(wsb + OB_LOGITS);
    float* sv     = (float*)(wsb + OB_SV);
    float* q      = (float*)(wsb + OB_Q);
    float* kh     = (float*)(wsb + OB_KH);
    float* gate   = (float*)(wsb + OB_GATE);
    float* delta  = (float*)(wsb + OB_DELTA);
    u16* p_bf   = (u16*)(wsb + OB_PBF);
    u16* vs_bf  = (u16*)(wsb + OB_VSBF);
    u16* g1_bf  = (u16*)(wsb + OB_G1BF);
    u16* d1_bf  = (u16*)(wsb + OB_D1BF);
    u16* fus_bf = (u16*)(wsb + OB_FUSBF);

    const int EW = (ROWS * HID + 255) / 256;
#define CVT(src, dst, srows, drows, scols, sld, dld) \
    k_cvt<<<((drows)*(dld) + 255) / 256, 256, 0, stream>>>(src, dst, srows, drows, scols, sld, dld)

    // ---- one-time pad/convert: activations entering the graph + all weights ----
    CVT(inputs, in_bf,  ROWS, ROWS, 322, 322, 352);
    CVT(hidden, hid_bf, ROWS, ROWS, 256, 256, 256);
    CVT(W1,  wbf + WE_W1,  256, 256, 322, 322, 352);
    CVT(Wih, wbf + WE_WIH, 768, 768, 256, 256, 256);
    CVT(Whh, wbf + WE_WHH, 768, 768, 256, 256, 256);
    CVT(Wp1, wbf + WE_WP1, 256, 256, 256, 256, 256);
    CVT(Wp2, wbf + WE_WP2, 33,  64,  256, 256, 256);
    CVT(Wv,  wbf + WE_WV,  32,  64,  29,  29,  32);
    CVT(Wq,  wbf + WE_WQ,  64,  64,  256, 256, 256);
    CVT(Wk,  wbf + WE_WKH, 64,  64,  256, 261, 256);   // Wk_h = Wk[:, :256]
    CVT(Wg1, wbf + WE_WG1, 256, 256, 288, 288, 288);
    CVT(Wd1, wbf + WE_WD1, 256, 256, 288, 288, 288);
    CVT(Wd2, wbf + WE_WD2, 27,  64,  256, 256, 256);
#undef CVT

    // 1. x = relu(inputs @ W1^T + b1)  -> bf16        (wide tile)
    launch_gemm(true, 128, x_bf, in_bf, wbf + WE_W1, b1, 352, HID, 352, 352, HID, stream);
    // 2-3. GRU gates -> f32                            (wide tile, N=768)
    launch_gemm(false, 128, gi, x_bf,   wbf + WE_WIH, bih, HID, 3 * HID, HID, HID, 3 * HID, stream);
    launch_gemm(false, 128, gh, hid_bf, wbf + WE_WHH, bhh, HID, 3 * HID, HID, HID, 3 * HID, stream);
    // 4. combine -> h (f32, into d_out) + h_bf (in_bf region dead now)
    k_gru<<<EW, 256, 0, stream>>>(gi, gh, hidden, h, h_bf);
    // 5. fusion_bf[:, :256] = h_bf   (gh region dead now)
    k_fuse_h<<<EW, 256, 0, stream>>>(h_bf, fus_bf);
    // 6-7. policy head
    launch_gemm(true, 128, p_bf,   h_bf, wbf + WE_WP1, bp1, HID, HID, HID, HID, HID, stream);
    launch_gemm(false, 64, logits, p_bf, wbf + WE_WP2, bp2, HID, NA,  HID, HID, 48,  stream);
    // 8. attack softmax -> value_source (bf16)
    k_value_source<<<(ROWS + 255) / 256, 256, 0, stream>>>(logits, vs_bf);
    // 9. sender_values = vs @ Wv^T + bv (K padded to 32) -> f32
    launch_gemm(false, 64, sv, vs_bf, wbf + WE_WV, bv, 32, 32, 32, 32, 32, stream);
    // 10-11. q / key_h projections -> f32
    launch_gemm(false, 64, q,  h_bf, wbf + WE_WQ,  bq,      HID, HEADS * HD, HID, HID, 64, stream);
    launch_gemm(false, 64, kh, h_bf, wbf + WE_WKH, nullptr, HID, HEADS * HD, HID, HID, 64, stream);
    // 12. attention core -> fusion_bf[:, 256:288]
    k_attn<<<(ROWS * HEADS) / 128, 128, 0, stream>>>(q, kh, rel, Wk, bk, sv,
                                                     nullk, nullv, lng, lnb, fus_bf);
    // 13-14. gate branch
    launch_gemm(true, 128, g1_bf, fus_bf, wbf + WE_WG1, bg1, 288, HID, 288, 288, HID, stream);
    k_gate<<<(ROWS + 255) / 256, 256, 0, stream>>>(g1_bf, Wg2, bg2, gate);
    // 15-16. delta branch
    launch_gemm(true, 128, d1_bf, fus_bf, wbf + WE_WD1, bd1, 288, HID, 288, 288, HID, stream);
    launch_gemm(false, 64, delta, d1_bf,  wbf + WE_WD2, bd2, HID, ATK, HID, HID, 32, stream);
    // 17. final fuse -> d_out[:ROWS*NA]
    k_final<<<(ROWS * NA + 255) / 256, 256, 0, stream>>>(logits, gate, delta, out);
}
